// MultiHeadAttention_89876485636820
// MI455X (gfx1250) — matrix operand order repaired and checked by reading, not compiled, not run on hardware
//
#include <hip/hip_runtime.h>

#define L_TOK    4096
#define C_DIM    512
#define HEADS    8
#define DK       64
#define BATCH    2
#define QKV_ROWS 1536

typedef __attribute__((ext_vector_type(16))) __bf16 v16bf;
typedef __attribute__((ext_vector_type(8)))  float  v8f;
typedef unsigned int u32x4 __attribute__((ext_vector_type(4)));
typedef int          i32x4 __attribute__((ext_vector_type(4)));
typedef int          i32x8 __attribute__((ext_vector_type(8)));

__device__ __forceinline__ unsigned short f2bf(float f) {
  unsigned int u = __builtin_bit_cast(unsigned int, f);
  u += 0x7FFFu + ((u >> 16) & 1u);  // round-to-nearest-even
  return (unsigned short)(u >> 16);
}

__device__ __forceinline__ v8f wmma_bf16(v16bf a, v16bf b, v8f c) {
  return __builtin_amdgcn_wmma_f32_16x16x32_bf16(false, a, false, b, (short)0, c,
                                                 false, false);
}

// A-frag from row-major [M][ld] (K contiguous). Two b128 loads:
// lane m=lane&15, hlf=lane>>4; K runs [8h..8h+7] and [16+8h..16+8h+7].
__device__ __forceinline__ v16bf load_a_frag(const unsigned short* base, int ld,
                                             int koff) {
  int lane = threadIdx.x & 31;
  int m = lane & 15, hlf = lane >> 4;
  union { v16bf v; u32x4 q[2]; } f;
  const unsigned short* p = base + m * ld + koff + 8 * hlf;
  f.q[0] = *(const u32x4*)(p);
  f.q[1] = *(const u32x4*)(p + 16);
  return f.v;
}

// B-frag from transposed storage [N][ld] (K contiguous). Two b128 loads:
// lane n=lane&15, hlf=lane>>4; K runs [8h..8h+7] and [16+8h..16+8h+7].
__device__ __forceinline__ v16bf load_bt_frag(const unsigned short* base, int ld,
                                              int koff) {
  int lane = threadIdx.x & 31;
  int n = lane & 15, hlf = lane >> 4;
  union { v16bf v; u32x4 q[2]; } f;
  const unsigned short* p = base + n * ld + koff + 8 * hlf;
  f.q[0] = *(const u32x4*)(p);
  f.q[1] = *(const u32x4*)(p + 16);
  return f.v;
}

// ---------------------------------------------------------------------------
// TDM: issue a 2-D tensor_load_to_lds. tile_d0 elements contiguous (dim0),
// tile_d1 rows with stride0 elements between rows. data_size = 2 bytes.
// ---------------------------------------------------------------------------
__device__ __forceinline__ void tdm_load_2d(const unsigned short* gptr,
                                            const unsigned short* lds_ptr,
                                            unsigned tile_d0, unsigned tile_d1,
                                            unsigned long long td0,
                                            unsigned long long td1,
                                            unsigned long long stride0) {
  unsigned long long ga = (unsigned long long)gptr;
  unsigned lds_off = (unsigned)(unsigned long long)lds_ptr;
  u32x4 g0;
  g0[0] = 1u;                                   // count=1 valid user descriptor
  g0[1] = lds_off;                              // lds_addr
  g0[2] = (unsigned)(ga & 0xFFFFFFFFu);         // global_addr[31:0]
  g0[3] = (unsigned)((ga >> 32) & 0x01FFFFFFu)  // global_addr[56:32]
          | (2u << 30);                         // type = 2 ("image")
  i32x8 g1;
  g1[0] = (int)(1u << 16);                                    // data_size=1 (2B)
  g1[1] = (int)((td0 & 0xFFFFu) << 16);                       // tensor_dim0 lo
  g1[2] = (int)(((td0 >> 16) & 0xFFFFu) | ((td1 & 0xFFFFu) << 16));
  g1[3] = (int)(((td1 >> 16) & 0xFFFFu) | (tile_d0 << 16));   // tile_dim0
  g1[4] = (int)(tile_d1 & 0xFFFFu);                           // tile_dim1 (d2=0)
  g1[5] = (int)(stride0 & 0xFFFFFFFFu);                       // dim0_stride lo
  g1[6] = (int)((stride0 >> 32) & 0xFFFFu);                   // dim0_stride hi
  g1[7] = 0;
  i32x4 gz = {0, 0, 0, 0};
#if defined(__clang_major__) && __clang_major__ >= 23
  i32x8 gz8 = {0, 0, 0, 0, 0, 0, 0, 0};
  __builtin_amdgcn_tensor_load_to_lds(g0, g1, gz, gz, gz8, 0);
#else
  __builtin_amdgcn_tensor_load_to_lds(g0, g1, gz, gz, 0);
#endif
}

// ---------------------------------------------------------------------------
// pack: fp32 -> bf16; fold SCALE into Wq,bq; fuse [WqS; Wkv]; x -> x^T [L][C]
// ---------------------------------------------------------------------------
__global__ void pack_kernel(const float* __restrict__ x,
                            const float* __restrict__ Wq,
                            const float* __restrict__ bq,
                            const float* __restrict__ Wkv,
                            const float* __restrict__ bkv,
                            const float* __restrict__ Wp,
                            unsigned short* __restrict__ Wqkv_b,
                            float* __restrict__ bqkv,
                            unsigned short* __restrict__ Wp_b,
                            unsigned short* __restrict__ xT) {
  const float scale = 0.125f;  // 1/sqrt(64)
  const long long NWQ  = (long long)C_DIM * C_DIM;
  const long long NWKV = 2LL * C_DIM * C_DIM;
  const long long NWP  = (long long)C_DIM * C_DIM;
  const long long NX   = (long long)BATCH * C_DIM * L_TOK;
  const long long TOT  = NWQ + NWKV + NWP + NX + QKV_ROWS;
  for (long long i = blockIdx.x * (long long)blockDim.x + threadIdx.x; i < TOT;
       i += (long long)gridDim.x * blockDim.x) {
    long long r = i;
    if (r < NWQ)  { Wqkv_b[r] = f2bf(Wq[r] * scale); continue; }
    r -= NWQ;
    if (r < NWKV) { Wqkv_b[NWQ + r] = f2bf(Wkv[r]); continue; }
    r -= NWKV;
    if (r < NWP)  { Wp_b[r] = f2bf(Wp[r]); continue; }
    r -= NWP;
    if (r < NX) {  // destination-indexed transpose: xT[b][l][c] = x[b][c][l]
      long long bb  = r / ((long long)L_TOK * C_DIM);
      long long rem = r % ((long long)L_TOK * C_DIM);
      long long l = rem / C_DIM, c = rem % C_DIM;
      xT[r] = f2bf(x[(bb * C_DIM + c) * L_TOK + l]);
      continue;
    }
    r -= NX;
    bqkv[r] = (r < C_DIM) ? bq[r] * scale : bkv[r - C_DIM];
  }
}

// ---------------------------------------------------------------------------
// GEMM: out[m,n] = sum_k A[m,k]*BT[n,k] + bias[m]; TDM double-buffered tiles.
// MODE 0: fp32 store [M][N] (final projection). MODE 1: QKV scatter stores.
// ---------------------------------------------------------------------------
template <int MODE>
__global__ void __launch_bounds__(256) gemm_bt_kernel(
    const unsigned short* __restrict__ A,   // [M][K] bf16
    const unsigned short* __restrict__ BT,  // [N][K] bf16 per batch
    const float* __restrict__ bias,         // [M]
    float* __restrict__ outF,               // MODE 0: [M][N] per batch
    unsigned short* __restrict__ qt,        // MODE 1: [B][H][L][DK]
    unsigned short* __restrict__ kt,        // MODE 1: [B][H][L][DK]
    unsigned short* __restrict__ vbuf,      // MODE 1: [B][C][L]
    int M, int N, int K, long long btStride, long long oStride) {
  __shared__ __align__(16) unsigned short As[2][128][32];
  __shared__ __align__(16) unsigned short Bs[2][128][32];

  const int tid = threadIdx.x;
  const int lane = tid & 31, wave = tid >> 5;
  const int wm = wave & 3, wn = wave >> 2;
  const int hlf = lane >> 4, nn = lane & 15;
  const int m0 = blockIdx.y * 128, n0 = blockIdx.x * 128;
  const int z = blockIdx.z;
  BT += (long long)z * btStride;

  v8f acc[2][4];
#pragma unroll
  for (int i = 0; i < 2; ++i)
#pragma unroll
    for (int j = 0; j < 4; ++j) acc[i][j] = (v8f)(0.0f);

  const int nsteps = K / 32;
  if (wave == 0) {  // preload tile 0 via TDM
    tdm_load_2d(A + (long long)m0 * K, &As[0][0][0], 32, 128, (unsigned)K,
                (unsigned)M, (unsigned)K);
    tdm_load_2d(BT + (long long)n0 * K, &Bs[0][0][0], 32, 128, (unsigned)K,
                (unsigned)N, (unsigned)K);
  }

  for (int ks = 0; ks < nsteps; ++ks) {
    const int buf = ks & 1;
    __syncthreads();  // everyone done reading buf^1 from previous iteration
    if (wave == 0) {
      if (ks + 1 < nsteps) {
        const int k1 = (ks + 1) * 32;
        tdm_load_2d(A + (long long)m0 * K + k1, &As[buf ^ 1][0][0], 32, 128,
                    (unsigned)K, (unsigned)M, (unsigned)K);
        tdm_load_2d(BT + (long long)n0 * K + k1, &Bs[buf ^ 1][0][0], 32, 128,
                    (unsigned)K, (unsigned)N, (unsigned)K);
        __builtin_amdgcn_s_wait_tensorcnt(2);  // current buf complete
      } else {
        __builtin_amdgcn_s_wait_tensorcnt(0);
      }
    }
    __syncthreads();

    v16bf af[2];
#pragma unroll
    for (int tm = 0; tm < 2; ++tm)
      af[tm] = load_a_frag(&As[buf][wm * 32 + tm * 16][0], 32, 0);
#pragma unroll
    for (int tn = 0; tn < 4; ++tn) {
      v16bf bfr = load_bt_frag(&Bs[buf][wn * 64 + tn * 16][0], 32, 0);
#pragma unroll
      for (int tm = 0; tm < 2; ++tm)
        acc[tm][tn] = wmma_bf16(af[tm], bfr, acc[tm][tn]);
    }
  }

#pragma unroll
  for (int tm = 0; tm < 2; ++tm)
#pragma unroll
    for (int tn = 0; tn < 4; ++tn)
#pragma unroll
      for (int j = 0; j < 8; ++j) {
        int m = m0 + wm * 32 + tm * 16 + j + 8 * hlf;
        int n = n0 + wn * 64 + tn * 16 + nn;
        float val = acc[tm][tn][j] + bias[m];
        if (MODE == 0) {
          outF[(long long)z * oStride + (long long)m * N + n] = val;
        } else {
          unsigned short hv = f2bf(val);
          if (m < C_DIM) {  // Q -> [b][h][t][d] (transposed per head)
            int hh = m >> 6, d = m & 63;
            qt[(((long long)z * HEADS + hh) * L_TOK + n) * DK + d] = hv;
          } else if (m < 2 * C_DIM) {  // K -> [b][h][s][d]
            int mm = m - C_DIM, hh = mm >> 6, d = mm & 63;
            kt[(((long long)z * HEADS + hh) * L_TOK + n) * DK + d] = hv;
          } else {  // V -> [b][c][t]
            vbuf[((long long)z * C_DIM + (m - 2 * C_DIM)) * L_TOK + n] = hv;
          }
        }
      }
}

// ---------------------------------------------------------------------------
// Flash attention: WG = (128 queries, head, batch); wave owns 16 queries.
// qt/kt: [B][H][L][64] bf16 (Q pre-scaled), vbuf: [B][C][L] bf16.
// K/V tiles staged by TDM, double-buffered. Writes O^T [B][L][C] bf16.
// ---------------------------------------------------------------------------
__global__ void __launch_bounds__(256) flash_attn_kernel(
    const unsigned short* __restrict__ qt, const unsigned short* __restrict__ kt,
    const unsigned short* __restrict__ vbuf, unsigned short* __restrict__ OT) {
  __shared__ __align__(16) unsigned short Ks[2][64][64];   // [s][d]
  __shared__ __align__(16) unsigned short Vs[2][64][64];   // [d][s]
  __shared__ __align__(16) unsigned short Ps[8][16][64];   // per-wave P

  const int tid = threadIdx.x;
  const int lane = tid & 31, wave = tid >> 5;
  const int hlf = lane >> 4, nn = lane & 15;
  const int t0 = blockIdx.x * 128;
  const int h = blockIdx.y;
  const int b = blockIdx.z;
  const int NIT = L_TOK / 64;

  const unsigned short* kbase = kt + ((long long)b * HEADS + h) * L_TOK * DK;
  const unsigned short* vbase = vbuf + ((long long)b * C_DIM + h * DK) * L_TOK;

  // Loop-invariant Q^T A-fragments straight from global (K-contiguous [t][d])
  const unsigned short* qrow =
      qt + (((long long)b * HEADS + h) * L_TOK + t0 + wave * 16) * DK;
  v16bf qa0 = load_a_frag(qrow, DK, 0);
  v16bf qa1 = load_a_frag(qrow, DK, 32);

  float m_r[8], l_r[8];
  v8f o_acc[4];
#pragma unroll
  for (int j = 0; j < 8; ++j) { m_r[j] = -1e30f; l_r[j] = 0.0f; }
#pragma unroll
  for (int dt = 0; dt < 4; ++dt) o_acc[dt] = (v8f)(0.0f);

  if (wave == 0) {  // preload block 0
    tdm_load_2d(kbase, &Ks[0][0][0], DK, 64, DK, L_TOK, DK);
    tdm_load_2d(vbase, &Vs[0][0][0], 64, DK, L_TOK, 1u << 20, L_TOK);
  }

  for (int it = 0; it < NIT; ++it) {
    const int buf = it & 1;
    __syncthreads();  // all waves done reading buf^1
    if (wave == 0) {
      if (it + 1 < NIT) {
        const long long s1 = (long long)(it + 1) * 64;
        tdm_load_2d(kbase + s1 * DK, &Ks[buf ^ 1][0][0], DK, 64, DK, L_TOK, DK);
        tdm_load_2d(vbase + s1, &Vs[buf ^ 1][0][0], 64, DK, L_TOK, 1u << 20,
                    L_TOK);
        __builtin_amdgcn_s_wait_tensorcnt(2);
      } else {
        __builtin_amdgcn_s_wait_tensorcnt(0);
      }
    }
    __syncthreads();

    // S = Q^T K : 16 queries x 64 keys, contraction over d (2 k-steps)
    v8f s_acc[4];
#pragma unroll
    for (int st = 0; st < 4; ++st) s_acc[st] = (v8f)(0.0f);
#pragma unroll
    for (int kk = 0; kk < 2; ++kk) {
      v16bf qa = (kk == 0) ? qa0 : qa1;
#pragma unroll
      for (int st = 0; st < 4; ++st) {
        v16bf bfr = load_bt_frag(&Ks[buf][st * 16][0], DK, kk * 32);
        s_acc[st] = wmma_bf16(qa, bfr, s_acc[st]);
      }
    }

    // Online softmax: row M = j + 8*hlf, row spread over the 16-lane group.
    float bm[8];
#pragma unroll
    for (int j = 0; j < 8; ++j)
      bm[j] = fmaxf(fmaxf(s_acc[0][j], s_acc[1][j]),
                    fmaxf(s_acc[2][j], s_acc[3][j]));
#pragma unroll
    for (int msk = 1; msk < 16; msk <<= 1)
#pragma unroll
      for (int j = 0; j < 8; ++j)
        bm[j] = fmaxf(bm[j], __shfl_xor(bm[j], msk, 32));

    float alpha[8], ls[8];
#pragma unroll
    for (int j = 0; j < 8; ++j) {
      float mn = fmaxf(m_r[j], bm[j]);
      alpha[j] = __expf(m_r[j] - mn);
      m_r[j] = mn;
      ls[j] = 0.0f;
    }
#pragma unroll
    for (int st = 0; st < 4; ++st)
#pragma unroll
      for (int j = 0; j < 8; ++j) {
        float p = __expf(s_acc[st][j] - m_r[j]);
        ls[j] += p;
        Ps[wave][j + 8 * hlf][st * 16 + nn] = f2bf(p);  // A-layout via LDS
      }
#pragma unroll
    for (int msk = 1; msk < 16; msk <<= 1)
#pragma unroll
      for (int j = 0; j < 8; ++j) ls[j] += __shfl_xor(ls[j], msk, 32);
#pragma unroll
    for (int j = 0; j < 8; ++j) l_r[j] = l_r[j] * alpha[j] + ls[j];
#pragma unroll
    for (int dt = 0; dt < 4; ++dt)
#pragma unroll
      for (int j = 0; j < 8; ++j) o_acc[dt][j] *= alpha[j];

    // O += P V^T : contraction over s (2 k-steps); same-wave LDS is in-order.
    v16bf pa0 = load_a_frag(&Ps[wave][0][0], 64, 0);
    v16bf pa1 = load_a_frag(&Ps[wave][0][0], 64, 32);
#pragma unroll
    for (int kk = 0; kk < 2; ++kk) {
      v16bf pa = (kk == 0) ? pa0 : pa1;
#pragma unroll
      for (int dt = 0; dt < 4; ++dt) {
        v16bf bfr = load_bt_frag(&Vs[buf][dt * 16][0], 64, kk * 32);
        o_acc[dt] = wmma_bf16(pa, bfr, o_acc[dt]);
      }
    }
  }

  // normalize; write O^T [B][L][C] (16 contiguous bf16 per store group)
  unsigned short* ob = OT + (long long)b * L_TOK * C_DIM;
  float rl[8];
#pragma unroll
  for (int j = 0; j < 8; ++j) rl[j] = 1.0f / l_r[j];
#pragma unroll
  for (int dt = 0; dt < 4; ++dt)
#pragma unroll
    for (int j = 0; j < 8; ++j) {
      int t = t0 + wave * 16 + j + 8 * hlf;
      int c = h * DK + dt * 16 + nn;
      ob[(long long)t * C_DIM + c] = f2bf(o_acc[dt][j] * rl[j]);
    }
}

// ---------------------------------------------------------------------------
extern "C" void kernel_launch(void* const* d_in, const int* in_sizes, int n_in,
                              void* d_out, int out_size, void* d_ws,
                              size_t ws_size, hipStream_t stream) {
  const float* x   = (const float*)d_in[0];
  const float* Wq  = (const float*)d_in[1];
  const float* bq  = (const float*)d_in[2];
  const float* Wkv = (const float*)d_in[3];
  const float* bkv = (const float*)d_in[4];
  const float* Wp  = (const float*)d_in[5];
  const float* bp  = (const float*)d_in[6];
  (void)in_sizes; (void)n_in; (void)out_size; (void)ws_size;

  char* ws = (char*)d_ws;
  size_t off = 0;
  unsigned short* Wqkv_b = (unsigned short*)(ws + off); off += (size_t)QKV_ROWS * C_DIM * 2;
  float* bqkv            = (float*)(ws + off);          off += (size_t)QKV_ROWS * 4;
  off = (off + 255) & ~(size_t)255;
  unsigned short* Wp_b = (unsigned short*)(ws + off); off += (size_t)C_DIM * C_DIM * 2;
  unsigned short* xT   = (unsigned short*)(ws + off); off += (size_t)BATCH * C_DIM * L_TOK * 2;
  unsigned short* qtb  = (unsigned short*)(ws + off); off += (size_t)BATCH * HEADS * L_TOK * DK * 2;
  unsigned short* ktb  = (unsigned short*)(ws + off); off += (size_t)BATCH * HEADS * L_TOK * DK * 2;
  unsigned short* vbuf = (unsigned short*)(ws + off); off += (size_t)BATCH * C_DIM * L_TOK * 2;
  unsigned short* OT   = (unsigned short*)(ws + off); off += (size_t)BATCH * L_TOK * C_DIM * 2;

  pack_kernel<<<2048, 256, 0, stream>>>(x, Wq, bq, Wkv, bkv, Wp, Wqkv_b, bqkv,
                                        Wp_b, xT);

  dim3 g1(L_TOK / 128, QKV_ROWS / 128, BATCH);
  gemm_bt_kernel<1><<<g1, 256, 0, stream>>>(
      Wqkv_b, xT, bqkv, nullptr, qtb, ktb, vbuf, QKV_ROWS, L_TOK, C_DIM,
      (long long)L_TOK * C_DIM, 0);

  dim3 g2(L_TOK / 128, HEADS, BATCH);
  flash_attn_kernel<<<g2, 256, 0, stream>>>(qtb, ktb, vbuf, OT);

  dim3 g3(L_TOK / 128, C_DIM / 128, BATCH);
  gemm_bt_kernel<0><<<g3, 256, 0, stream>>>(
      Wp_b, OT, bp, (float*)d_out, nullptr, nullptr, nullptr, C_DIM, L_TOK,
      C_DIM, (long long)L_TOK * C_DIM, (long long)C_DIM * L_TOK);
}
